// ReverseEmbedding_43422119362854
// MI455X (gfx1250) — compile-verified, matrix operand order checked by reading, not run on hardware
//
#include <hip/hip_runtime.h>

typedef __attribute__((ext_vector_type(16))) __bf16         v16bf;
typedef __attribute__((ext_vector_type(8)))  float          v8f;
typedef __attribute__((ext_vector_type(4)))  float          f32x4;
typedef __attribute__((ext_vector_type(4)))  unsigned int   u32x4;
typedef __attribute__((ext_vector_type(4)))  unsigned short u16x4;

#define N_TOK  8192
#define DIM    1024
#define VOCAB  32000
#define MB     64          // rows per block
#define NSUB   4           // 16-row WMMA subtiles per block
#define NV     2           // v-tiles per wave pass: halves LDS A-traffic per WMMA
#define LDSROW 1032        // 1024 + 8 bf16 pad (16B-aligned rows, breaks bank aliasing)
#define KC     64          // k-chunk (2 WMMA k-steps); ko loop stays DYNAMIC

union Frag {
    v16bf v;
    u32x4 u[2];
};

__device__ __forceinline__ void loadFrag(Frag& f, const unsigned short* p) {
    f.u[0] = *(const u32x4*)(p);        // K +0..7  (half 0) / +8..15  (half 1)
    f.u[1] = *(const u32x4*)(p + 16);   // K +16..23 (half 0) / +24..31 (half 1)
}

__device__ __forceinline__ unsigned short f2bf(float x) {
    unsigned b = __float_as_uint(x);
    unsigned r = ((b >> 16) & 1u) + 0x7FFFu;   // round-to-nearest-even
    return (unsigned short)((b + r) >> 16);
}

// ---------------------------------------------------------------- zero output
__global__ void zero_out_kernel(f32x4* __restrict__ out, long long n4) {
    long long i      = (long long)blockIdx.x * blockDim.x + threadIdx.x;
    long long stride = (long long)gridDim.x * blockDim.x;
    f32x4 z = {0.f, 0.f, 0.f, 0.f};
    for (; i < n4; i += stride)
        __builtin_nontemporal_store(z, out + i);   // keep the 1GB stream out of L2
}

// ------------------------------------------------------- convert inputs -> bf16
__global__ void convert_a_kernel(const float* __restrict__ A,
                                 unsigned short* __restrict__ A16, int n4) {
    int i = blockIdx.x * blockDim.x + threadIdx.x;
    if (i >= n4) return;
    f32x4 f = ((const f32x4*)A)[i];
    u16x4 o;
    o.x = f2bf(f.x); o.y = f2bf(f.y); o.z = f2bf(f.z); o.w = f2bf(f.w);
    ((u16x4*)A16)[i] = o;
}

// ---------------------- normalize each embedding row, emit bf16 (fold 1/||E_v||)
__global__ void norm_e_kernel(const float* __restrict__ E,
                              unsigned short* __restrict__ E16) {
    __shared__ float red[8];
    const int v = blockIdx.x;
    const float* row = E + (size_t)v * DIM;
    float s = 0.f;
    for (int i = threadIdx.x; i < DIM; i += 256) {
        float x = row[i];
        s += x * x;
    }
#pragma unroll
    for (int off = 16; off > 0; off >>= 1)
        s += __shfl_xor(s, off, 32);               // wave32
    const int lane = threadIdx.x & 31;
    const int wave = threadIdx.x >> 5;
    if (lane == 0) red[wave] = s;
    __syncthreads();
    if (threadIdx.x == 0) {
        float t = 0.f;
#pragma unroll
        for (int w = 0; w < 8; ++w) t += red[w];
        red[0] = rsqrtf(t);
    }
    __syncthreads();
    const float r = red[0];
    for (int i = threadIdx.x; i < DIM; i += 256)
        E16[(size_t)v * DIM + i] = f2bf(row[i] * r);
}

// --------------------------- fused bf16 WMMA GEMM + running argmax + scatter 1.0
__global__ __launch_bounds__(256)
__attribute__((amdgpu_waves_per_eu(2)))   // cap at 256 VGPRs: no MSB regime
void gemm_argmax_kernel(const unsigned short* __restrict__ A16,
                        const unsigned short* __restrict__ E16,
                        float* __restrict__ out) {
    extern __shared__ char smem_raw[];
    unsigned short* As = (unsigned short*)smem_raw;                       // MB x LDSROW
    unsigned long long* bestPack =
        (unsigned long long*)(smem_raw + (size_t)MB * LDSROW * sizeof(unsigned short));

    const int tid = threadIdx.x;
    const int n0  = blockIdx.x * MB;

    // Stage the 64x1024 bf16 A tile into LDS (16B chunks, conflict-free pattern).
    for (int i = tid; i < MB * (DIM / 8); i += 256) {
        int row = i >> 7;          // i / 128
        int c   = i & 127;
        u32x4 d = *(const u32x4*)(A16 + (size_t)(n0 + row) * DIM + c * 8);
        *(u32x4*)(As + row * LDSROW + c * 8) = d;
    }
    if (tid < MB) bestPack[tid] = 0ull;
    __syncthreads();

    const int wave = tid >> 5;
    const int lane = tid & 31;
    const int half = lane >> 4;    // K-half per ISA fragment layout
    const int l4   = lane & 15;

    // Integer offsets into the __shared__ base: every A access is a GEP off
    // `As`, so address-space inference yields ds_load_b128 (NOT flat_load),
    // and the dynamic `ko` below keeps LICM from hoisting/spilling them.
    int aOff[NSUB];
#pragma unroll
    for (int s = 0; s < NSUB; ++s)
        aOff[s] = (s * 16 + l4) * LDSROW + half * 8;

    float bestV[NSUB][8];
    int   bestI[NSUB][8];
#pragma unroll
    for (int s = 0; s < NSUB; ++s)
#pragma unroll
        for (int j = 0; j < 8; ++j) { bestV[s][j] = -3.402823e38f; bestI[s][j] = 0; }

    const v8f zeroAcc = {0.f, 0.f, 0.f, 0.f, 0.f, 0.f, 0.f, 0.f};
    const int NPASS = 16 * NV;                        // 32 vocab rows per pass

    // Each wave owns passes wave, wave+8, ... : block streams E exactly once.
    for (int vp = wave; vp < VOCAB / NPASS; vp += 8) {
        const int v0 = vp * NPASS;
        const unsigned short* Bp = E16 + (size_t)(v0 + l4) * DIM + half * 8;

        // Warm the cache with the next pass's embedding rows.
        if (vp + 8 < VOCAB / NPASS) {
            __builtin_prefetch(E16 + (size_t)(v0 + 8 * NPASS + l4) * DIM + half * 8, 0, 3);
            __builtin_prefetch(E16 + (size_t)(v0 + 8 * NPASS + 16 + l4) * DIM + half * 8, 0, 3);
        }

        v8f acc0[NSUB], acc1[NSUB];
#pragma unroll
        for (int s = 0; s < NSUB; ++s) { acc0[s] = zeroAcc; acc1[s] = zeroAcc; }

        // DYNAMIC chunk loop (unroll forbidden) with a 2-step unrolled body:
        // enough lookahead to overlap L2/LDS latency, small enough to not spill.
#pragma unroll 1
        for (int ko = 0; ko < DIM; ko += KC) {
            // cover the loop-backedge latency for the B stream
            __builtin_prefetch(Bp + ko + KC, 0, 3);
            __builtin_prefetch(Bp + 16 * DIM + ko + KC, 0, 3);
#pragma unroll
            for (int kk = 0; kk < KC / 32; ++kk) {
                const int k = ko + kk * 32;
                Frag b0, b1;
                loadFrag(b0, Bp + k);                 // v-tile 0
                loadFrag(b1, Bp + k + 16 * DIM);      // v-tile 1: +32KB immediate
                Frag a[NSUB];
#pragma unroll
                for (int s = 0; s < NSUB; ++s)
                    loadFrag(a[s], As + (aOff[s] + k));
#pragma unroll
                for (int s = 0; s < NSUB; ++s)
                    acc0[s] = __builtin_amdgcn_wmma_f32_16x16x32_bf16(
                        false, a[s].v, false, b0.v, (short)0, acc0[s], false, false);
#pragma unroll
                for (int s = 0; s < NSUB; ++s)
                    acc1[s] = __builtin_amdgcn_wmma_f32_16x16x32_bf16(
                        false, a[s].v, false, b1.v, (short)0, acc1[s], false, false);
            }
        }

        const int cand0 = v0 + l4;                    // C layout: N = lane & 15
        const int cand1 = v0 + 16 + l4;
#pragma unroll
        for (int s = 0; s < NSUB; ++s)
#pragma unroll
            for (int j = 0; j < 8; ++j) {
                float x0 = acc0[s][j];
                if (x0 > bestV[s][j]) { bestV[s][j] = x0; bestI[s][j] = cand0; }
                float x1 = acc1[s][j];
                if (x1 > bestV[s][j]) { bestV[s][j] = x1; bestI[s][j] = cand1; }
            }
    }

    // Reduce argmax across the 16 lanes of each half, then across waves via LDS.
#pragma unroll
    for (int s = 0; s < NSUB; ++s) {
#pragma unroll
        for (int j = 0; j < 8; ++j) {
            unsigned u = __float_as_uint(bestV[s][j]);
            u = (u & 0x80000000u) ? ~u : (u | 0x80000000u);   // order-preserving
            unsigned long long key =
                ((unsigned long long)u << 32) | (unsigned)(~bestI[s][j]); // tie -> min idx
#pragma unroll
            for (int off = 8; off > 0; off >>= 1) {
                unsigned long long o = __shfl_xor(key, off, 32);
                if (o > key) key = o;
            }
            if (l4 == 0)
                atomicMax(&bestPack[s * 16 + half * 8 + j], key);  // ds_max_u64
        }
    }
    __syncthreads();

    if (tid < MB) {
        unsigned long long key = bestPack[tid];
        int idx = (int)(~(unsigned)(key & 0xFFFFFFFFull));
        __builtin_nontemporal_store(1.0f, out + (size_t)(n0 + tid) * VOCAB + idx);
    }
}

// -------------------------------------------------------------------- launcher
extern "C" void kernel_launch(void* const* d_in, const int* in_sizes, int n_in,
                              void* d_out, int out_size, void* d_ws, size_t ws_size,
                              hipStream_t stream) {
    const float* A = (const float*)d_in[0];   // [8192, 1024] f32
    const float* E = (const float*)d_in[1];   // [32000, 1024] f32
    float* out = (float*)d_out;               // [8192, 32000] f32 one-hot

    unsigned short* A16 = (unsigned short*)d_ws;                              // 16 MB
    unsigned short* E16 =
        (unsigned short*)((char*)d_ws + (size_t)N_TOK * DIM * sizeof(unsigned short)); // +16 MB

    // 1) zero the one-hot output (non-temporal, before E16 populates L2)
    long long n4 = (long long)out_size / 4;
    zero_out_kernel<<<8192, 256, 0, stream>>>((f32x4*)out, n4);

    // 2) inputs -> bf16
    int n4a = N_TOK * DIM / 4;
    convert_a_kernel<<<n4a / 256, 256, 0, stream>>>(A, A16, n4a);

    // 3) embedding rows -> normalized bf16 (L2-resident 64 MB)
    norm_e_kernel<<<VOCAB, 256, 0, stream>>>(E, E16);

    // 4) fused WMMA GEMM + argmax + scatter
    size_t shmem = (size_t)MB * LDSROW * sizeof(unsigned short) +
                   (size_t)MB * sizeof(unsigned long long);
    gemm_argmax_kernel<<<N_TOK / MB, 256, shmem, stream>>>(A16, E16, out);
}
